// MolGAT_7241314861280
// MI455X (gfx1250) — compile-verified, matrix operand order checked by reading, not run on hardware
//
#include <hip/hip_runtime.h>
#include <hip/hip_bf16.h>
#include <math.h>

typedef __attribute__((ext_vector_type(16))) _Float16 v16h;
typedef __attribute__((ext_vector_type(8)))  _Float16 v8h;
typedef __attribute__((ext_vector_type(8)))  float    v8f;

#define N_NODES 30000
#define N_EDGES 300000
#define EDGES_P (N_EDGES + N_NODES)   // + self loops
#define G_GRAPHS 1024
#define HID 256
#define NHEAD 8
#define CDIM 32

// ---------- helpers ----------
__device__ __forceinline__ void atomicMaxF(float* addr, float val) {
  // valid given init to -inf; positive floats compare as ints, negative as uints (reversed)
  if (val >= 0.0f) atomicMax((int*)addr, __float_as_int(val));
  else             atomicMin((unsigned int*)addr, __float_as_uint(val));
}

__global__ void fill_f32(float* __restrict__ p, float v, int n) {
  int i = blockIdx.x * blockDim.x + threadIdx.x;
  if (i < n) p[i] = v;
}

// ---------- W (f32, [256,256]) -> f16 ----------
__global__ void conv_w_f16(const float* __restrict__ w, _Float16* __restrict__ wh) {
  int i = blockIdx.x * blockDim.x + threadIdx.x;   // 65536 threads
  wh[i] = (_Float16)w[i];
}

// ---------- embedding lookup with max_norm=1 -> f16 features ----------
__global__ void embed_norm(const int* __restrict__ ids, const float* __restrict__ emb,
                           _Float16* __restrict__ xh) {
  __shared__ float red[256];
  __shared__ float s_scale;
  int n = blockIdx.x, t = threadIdx.x;
  float v = emb[(size_t)ids[n] * HID + t];
  red[t] = v * v;
  __syncthreads();
  for (int off = 128; off > 0; off >>= 1) {
    if (t < off) red[t] += red[t + off];
    __syncthreads();
  }
  if (t == 0) {
    float sc = 1.0f / (sqrtf(red[0]) + 1e-12f);
    s_scale = sc < 1.0f ? sc : 1.0f;
  }
  __syncthreads();
  xh[(size_t)n * HID + t] = (_Float16)(v * s_scale);
}

// ---------- H = X * W^T  via v_wmma_f32_16x16x32_f16 ----------
// One wave computes a 16x16 output tile; K = 256 (8 WMMA steps).
// X: [30000, 256] f16 row-major, Wh: [256, 256] f16 row-major (rows = output cols).
__global__ void gemm_wmma_f16(const _Float16* __restrict__ X,
                              const _Float16* __restrict__ Wh,
                              float* __restrict__ Hout) {
  const int K = 256;
  int wave = (blockIdx.x * blockDim.x + threadIdx.x) >> 5;  // 0..29999
  int lane = threadIdx.x & 31;
  int rowTile = wave >> 4;   // 0..1874  (30000/16 rows of tiles)
  int colTile = wave & 15;   // 0..15    (256/16 cols of tiles)

  int row  = rowTile * 16 + (lane & 15);   // A: M index per lane
  int wrow = colTile * 16 + (lane & 15);   // B: N index per lane (= row of W)
  int kb   = (lane >> 4) << 3;             // 0 or 8: lane-half K base

  const _Float16* xp = X  + (size_t)row  * K + kb;
  const _Float16* wp = Wh + (size_t)wrow * K + kb;

  v8f acc = {};
#pragma unroll
  for (int k0 = 0; k0 < K; k0 += 32) {
    v8h alo = *(const v8h*)(xp + k0);        // K = kb .. kb+7
    v8h ahi = *(const v8h*)(xp + k0 + 16);   // K = kb+16 .. kb+23
    v8h blo = *(const v8h*)(wp + k0);
    v8h bhi = *(const v8h*)(wp + k0 + 16);
    v16h a = __builtin_shufflevector(alo, ahi, 0,1,2,3,4,5,6,7,8,9,10,11,12,13,14,15);
    v16h b = __builtin_shufflevector(blo, bhi, 0,1,2,3,4,5,6,7,8,9,10,11,12,13,14,15);
    acc = __builtin_amdgcn_wmma_f32_16x16x32_f16(false, a, false, b,
                                                 (short)0, acc, false, false);
  }

  // C layout: VGPR r holds (M = r + 8*(lane>>4), N = lane&15)
  int mbase = rowTile * 16 + ((lane >> 4) << 3);
  int ncol  = colTile * 16 + (lane & 15);
#pragma unroll
  for (int r = 0; r < 8; ++r)
    Hout[(size_t)(mbase + r) * HID + ncol] = acc[r];
}

// ---------- e_src / e_dst:  [N, H] = einsum(h[n,h,c], a[h,c]) ----------
__global__ void attn_logits(const float* __restrict__ H32,
                            const float* __restrict__ a_src,
                            const float* __restrict__ a_dst,
                            float* __restrict__ esrc, float* __restrict__ edst) {
  int t = blockIdx.x * blockDim.x + threadIdx.x;
  if (t >= N_NODES * NHEAD) return;
  int n = t >> 3, hh = t & 7;
  const float* hp = H32 + (size_t)n * HID + hh * CDIM;
  const float* ap = a_src + hh * CDIM;
  const float* bp = a_dst + hh * CDIM;
  float s1 = 0.f, s2 = 0.f;
#pragma unroll
  for (int c = 0; c < CDIM; ++c) { float hv = hp[c]; s1 += hv * ap[c]; s2 += hv * bp[c]; }
  esrc[t] = s1;
  edst[t] = s2;
}

__device__ __forceinline__ void edge_endpoints(const int* __restrict__ ei, int e,
                                               int& src, int& dst) {
  if (e < N_EDGES) { src = ei[e]; dst = ei[N_EDGES + e]; }
  else             { src = dst = e - N_EDGES; }   // self-loop
}

// ---------- pass 1: segment max of leaky_relu logits ----------
__global__ void edge_max(const int* __restrict__ ei, const float* __restrict__ esrc,
                         const float* __restrict__ edst, float* __restrict__ m) {
  int e = blockIdx.x * blockDim.x + threadIdx.x;
  if (e >= EDGES_P) return;
  int src, dst; edge_endpoints(ei, e, src, dst);
#pragma unroll
  for (int hh = 0; hh < NHEAD; ++hh) {
    float v = esrc[src * NHEAD + hh] + edst[dst * NHEAD + hh];
    v = v > 0.f ? v : 0.2f * v;
    atomicMaxF(&m[dst * NHEAD + hh], v);
  }
}

// ---------- pass 2: segment sum of exp(e - m) ----------
__global__ void edge_sum(const int* __restrict__ ei, const float* __restrict__ esrc,
                         const float* __restrict__ edst, const float* __restrict__ m,
                         float* __restrict__ s) {
  int e = blockIdx.x * blockDim.x + threadIdx.x;
  if (e >= EDGES_P) return;
  int src, dst; edge_endpoints(ei, e, src, dst);
#pragma unroll
  for (int hh = 0; hh < NHEAD; ++hh) {
    float v = esrc[src * NHEAD + hh] + edst[dst * NHEAD + hh];
    v = v > 0.f ? v : 0.2f * v;
    atomicAdd(&s[dst * NHEAD + hh], __expf(v - m[dst * NHEAD + hh]));
  }
}

// ---------- pass 3: agg[dst] += alpha * h[src]  (one wave32 per edge) ----------
__global__ void edge_aggregate(const int* __restrict__ ei, const float* __restrict__ esrc,
                               const float* __restrict__ edst, const float* __restrict__ m,
                               const float* __restrict__ s, const float* __restrict__ H32,
                               float* __restrict__ agg) {
  int wid  = (blockIdx.x * blockDim.x + threadIdx.x) >> 5;
  int lane = threadIdx.x & 31;
  if (wid >= EDGES_P) return;
  int src, dst; edge_endpoints(ei, wid, src, dst);
  const float* hs = H32 + (size_t)src * HID;
  float*       ag = agg + (size_t)dst * HID;
#pragma unroll
  for (int hh = 0; hh < NHEAD; ++hh) {
    float v = esrc[src * NHEAD + hh] + edst[dst * NHEAD + hh];
    v = v > 0.f ? v : 0.2f * v;
    float ex    = __expf(v - m[dst * NHEAD + hh]);
    float alpha = ex / (s[dst * NHEAD + hh] + 1e-16f);
    atomicAdd(&ag[hh * CDIM + lane], hs[hh * CDIM + lane] * alpha);
  }
}

// ---------- x = relu(agg + b), emit f32 (for pooling) and f16 (next GEMM) ----------
__global__ void bias_relu(const float* __restrict__ agg, const float* __restrict__ b,
                          float* __restrict__ x32, _Float16* __restrict__ xh) {
  size_t i = (size_t)blockIdx.x * 256 + threadIdx.x;  // N*256 threads
  float v = agg[i] + b[i & 255];
  v = fmaxf(v, 0.f);
  x32[i] = v;
  xh[i]  = (_Float16)v;
}

// ---------- per-graph feature max-pool ----------
__global__ void pool_max(const int* __restrict__ batch, const float* __restrict__ x32,
                         float* __restrict__ pool) {
  size_t i = (size_t)blockIdx.x * 256 + threadIdx.x;  // N*256 threads
  int n = (int)(i >> 8), o = (int)(i & 255);
  atomicMaxF(&pool[(size_t)batch[n] * HID + o], x32[i]);
}

// ---------- out[g] = dot(pool[g], fw) + fb ----------
__global__ void final_dot(const float* __restrict__ pool, const float* __restrict__ fw,
                          const float* __restrict__ fb, float* __restrict__ out) {
  __shared__ float red[256];
  int g = blockIdx.x, t = threadIdx.x;
  red[t] = pool[(size_t)g * HID + t] * fw[t];
  __syncthreads();
  for (int off = 128; off > 0; off >>= 1) {
    if (t < off) red[t] += red[t + off];
    __syncthreads();
  }
  if (t == 0) out[g] = red[0] + fb[0];
}

extern "C" void kernel_launch(void* const* d_in, const int* in_sizes, int n_in,
                              void* d_out, int out_size, void* d_ws, size_t ws_size,
                              hipStream_t stream) {
  const int*   x_ids = (const int*)d_in[0];
  const int*   ei    = (const int*)d_in[1];
  const int*   batch = (const int*)d_in[2];
  const float* emb   = (const float*)d_in[3];
  const float* fw    = (const float*)d_in[4];
  const float* fb    = (const float*)d_in[5];

  char* ws = (char*)d_ws;
  size_t off = 0;
  auto alloc = [&](size_t bytes) -> char* {
    char* p = ws + off;
    off = (off + bytes + 255) & ~(size_t)255;
    return p;
  };
  _Float16* XH   = (_Float16*)alloc((size_t)N_NODES * HID * sizeof(_Float16));
  float*    X32  = (float*)   alloc((size_t)N_NODES * HID * sizeof(float));
  float*    H32  = (float*)   alloc((size_t)N_NODES * HID * sizeof(float));
  float*    AGG  = (float*)   alloc((size_t)N_NODES * HID * sizeof(float));
  _Float16* WH   = (_Float16*)alloc((size_t)HID * HID * sizeof(_Float16));
  float*    ES   = (float*)   alloc((size_t)N_NODES * NHEAD * sizeof(float));
  float*    ED   = (float*)   alloc((size_t)N_NODES * NHEAD * sizeof(float));
  float*    MB   = (float*)   alloc((size_t)N_NODES * NHEAD * sizeof(float));
  float*    SB   = (float*)   alloc((size_t)N_NODES * NHEAD * sizeof(float));
  float*    POOL = (float*)   alloc((size_t)G_GRAPHS * HID * sizeof(float));

  const int NHn   = N_NODES * NHEAD;          // 240000
  const int nfeat = N_NODES * HID;            // 7,680,000 (divisible by 256)

  // embedding (+ max_norm) -> f16 features
  embed_norm<<<N_NODES, 256, 0, stream>>>(x_ids, emb, XH);

  for (int l = 0; l < 5; ++l) {
    const float* W  = (const float*)d_in[6 + 4 * l];
    const float* as = (const float*)d_in[7 + 4 * l];
    const float* ad = (const float*)d_in[8 + 4 * l];
    const float* b  = (const float*)d_in[9 + 4 * l];

    conv_w_f16<<<(HID * HID) / 256, 256, 0, stream>>>(W, WH);
    // 30000 waves (1875 row-tiles x 16 col-tiles), 8 waves/block
    gemm_wmma_f16<<<3750, 256, 0, stream>>>(XH, WH, H32);

    attn_logits<<<(NHn + 255) / 256, 256, 0, stream>>>(H32, as, ad, ES, ED);

    fill_f32<<<(NHn + 255) / 256, 256, 0, stream>>>(MB, -INFINITY, NHn);
    fill_f32<<<(NHn + 255) / 256, 256, 0, stream>>>(SB, 0.0f, NHn);
    fill_f32<<<nfeat / 256, 256, 0, stream>>>(AGG, 0.0f, nfeat);

    edge_max<<<(EDGES_P + 255) / 256, 256, 0, stream>>>(ei, ES, ED, MB);
    edge_sum<<<(EDGES_P + 255) / 256, 256, 0, stream>>>(ei, ES, ED, MB, SB);
    // one wave32 per edge, 8 waves per block
    edge_aggregate<<<(EDGES_P + 7) / 8, 256, 0, stream>>>(ei, ES, ED, MB, SB, H32, AGG);

    bias_relu<<<nfeat / 256, 256, 0, stream>>>(AGG, b, X32, XH);
  }

  fill_f32<<<(G_GRAPHS * HID) / 256, 256, 0, stream>>>(POOL, -INFINITY, G_GRAPHS * HID);
  pool_max<<<nfeat / 256, 256, 0, stream>>>(batch, X32, POOL);
  final_dot<<<G_GRAPHS, 256, 0, stream>>>(POOL, fw, fb, (float*)d_out);
}
